// EvoMultiheadSelfAttention_47674136985903
// MI455X (gfx1250) — matrix operand order repaired and checked by reading, not compiled, not run on hardware
//
#include <hip/hip_runtime.h>
#include <hip/hip_bf16.h>
#include <cstdint>

// EvoMultiheadSelfAttention for MI455X (gfx1250), wave32 + WMMA f16.
// B=2, T=2048, DM=1024, H=16, DH=64, window=64.
// Workspace usage: ~81 MB (f16 x/weights/q/k/v/attn + f32 full/window outputs).

#define TSEQ 2048
#define NB   2
#define NH   16
#define DHD  64
#define DMOD 1024
#define WIN  64

typedef __attribute__((ext_vector_type(16))) _Float16 v16h;
typedef __attribute__((ext_vector_type(8)))  _Float16 v8h;
typedef __attribute__((ext_vector_type(8)))  float    v8f;

#if defined(__has_builtin)
#if __has_builtin(__builtin_amdgcn_global_load_async_to_lds_b128)
#define HAVE_ASYNC_LDS 1
#endif
#endif
#ifndef HAVE_ASYNC_LDS
#define HAVE_ASYNC_LDS 0
#endif

__device__ __forceinline__ v16h cat16(v8h lo, v8h hi) {
  return __builtin_shufflevector(lo, hi, 0,1,2,3,4,5,6,7,8,9,10,11,12,13,14,15);
}
// Contiguous 16-halfword load (B-operand layout: lane group picks K-chunk).
__device__ __forceinline__ v16h ldB(const _Float16* p) {
  return cat16(*(const v8h*)(p), *(const v8h*)(p + 8));
}
// A-operand layout (contiguous, matches ldB): lane<16 holds K{0..15},
// lane>=16 holds K{16..31}. p = row base of current 32-wide K chunk.
__device__ __forceinline__ v16h ldA(const _Float16* p, int lh) {
  return cat16(*(const v8h*)(p + lh * 16), *(const v8h*)(p + lh * 16 + 8));
}
__device__ __forceinline__ void st16(_Float16* p, v16h v) {
  *(v8h*)(p)     = __builtin_shufflevector(v, v, 0,1,2,3,4,5,6,7);
  *(v8h*)(p + 8) = __builtin_shufflevector(v, v, 8,9,10,11,12,13,14,15);
}
__device__ __forceinline__ v8f wmma16(v16h a, v16h b, v8f c) {
  return __builtin_amdgcn_wmma_f32_16x16x32_f16(false, a, false, b, (short)0, c,
                                                false, false);
}

// 32-byte global -> LDS copy, async on CDNA5 (tracked by ASYNCcnt).
#if HAVE_ASYNC_LDS
typedef int v4i_ __attribute__((vector_size(16)));
typedef __attribute__((address_space(1))) v4i_* gv4p;
typedef __attribute__((address_space(3))) v4i_* lv4p;
__device__ __forceinline__ void cp_async32(const _Float16* g, _Float16* l) {
  gv4p gp = (gv4p)(uintptr_t)g;                 // generic == global flat addr
  lv4p lp = (lv4p)(uint32_t)(uintptr_t)l;       // low 32 bits = LDS offset
  __builtin_amdgcn_global_load_async_to_lds_b128(gp, lp, 0, 0);
  __builtin_amdgcn_global_load_async_to_lds_b128(gp, lp, 16, 0);
}
__device__ __forceinline__ void cp_wait() {
#if __has_builtin(__builtin_amdgcn_s_wait_asynccnt)
  __builtin_amdgcn_s_wait_asynccnt(0);
#else
  asm volatile("s_wait_asynccnt 0x0" ::: "memory");
#endif
}
#else
__device__ __forceinline__ void cp_async32(const _Float16* g, _Float16* l) {
  st16(l, ldB(g));
}
__device__ __forceinline__ void cp_wait() {}
#endif

// ---------------------------------------------------------------- convert
__global__ void cvt_f32_f16(const float* __restrict__ src,
                            _Float16* __restrict__ dst, int n) {
  int i = blockIdx.x * blockDim.x + threadIdx.x;
  if (i < n) dst[i] = (_Float16)src[i];
}

// ---------------------------------------------------------------- GEMM
// C[M,N] = A[M,K](f16,rowmajor) * Bt[N,K](f16,rowmajor)^T + bias.
// mode 0: outF[M,N] f32.  mode 1: head-scatter f16 -> [b][h][t][dh] * scale.
// Double-buffered LDS; tile k+1 streams in via async copies while tile k
// feeds the WMMA pipe. All frags hoisted so the 8 WMMAs issue back-to-back
// behind a single s_wait_dscnt.
__global__ __launch_bounds__(256) void gemm_wmma(
    const _Float16* __restrict__ A, const _Float16* __restrict__ Bt,
    const float* __restrict__ bias, float* __restrict__ outF,
    _Float16* __restrict__ outH, int M, int N, int K, int mode, float scale) {
  __shared__ _Float16 As[2 * 128 * 40];
  __shared__ _Float16 Bs[2 * 128 * 40];
  const int tid  = threadIdx.x;
  const int lane = tid & 31;
  const int wave = tid >> 5;
  const int ln = lane & 15, lh = lane >> 4;
  const int wm = (wave >> 2) * 64;   // 2 waves in M
  const int wn = (wave & 3) * 32;    // 4 waves in N
  const int m0 = blockIdx.y * 128;
  const int n0 = blockIdx.x * 128;
  const int lrow = tid >> 1;
  const int loff = (tid & 1) * 16;

  const _Float16* gA = &A[(size_t)(m0 + lrow) * K + loff];
  const _Float16* gB = &Bt[(size_t)(n0 + lrow) * K + loff];
  _Float16* lA = &As[lrow * 40 + loff];
  _Float16* lB = &Bs[lrow * 40 + loff];

  // prologue: fill buffer 0
  cp_async32(gA, lA);
  cp_async32(gB, lB);
  cp_wait();
  __syncthreads();

  v8f acc[4][2] = {};
  int buf = 0;
  for (int k0 = 0; k0 < K; k0 += 32) {
    const int nxt = k0 + 32;
    if (nxt < K) {  // stream next slab into the other buffer
      cp_async32(gA + nxt, lA + (buf ^ 1) * 5120);
      cp_async32(gB + nxt, lB + (buf ^ 1) * 5120);
    }
    if (k0 + 64 < K) {  // warm GL2 two slabs ahead
      __builtin_prefetch(gA + k0 + 64, 0, 1);
      __builtin_prefetch(gB + k0 + 64, 0, 1);
    }
    const _Float16* Ab = &As[buf * 5120];
    const _Float16* Bb = &Bs[buf * 5120];
    v16h bf[2], af[4];
#pragma unroll
    for (int j = 0; j < 2; ++j)
      bf[j] = ldB(&Bb[(wn + j * 16 + ln) * 40 + lh * 16]);
#pragma unroll
    for (int i = 0; i < 4; ++i)
      af[i] = ldA(&Ab[(wm + i * 16 + ln) * 40], lh);
#pragma unroll
    for (int i = 0; i < 4; ++i)
#pragma unroll
      for (int j = 0; j < 2; ++j) acc[i][j] = wmma16(af[i], bf[j], acc[i][j]);
    if (nxt < K) {
      cp_wait();
      __syncthreads();
      buf ^= 1;
    }
  }
#pragma unroll
  for (int i = 0; i < 4; ++i)
#pragma unroll
    for (int j = 0; j < 2; ++j)
#pragma unroll
      for (int r = 0; r < 8; ++r) {
        int gm = m0 + wm + i * 16 + r + 8 * lh;
        int gn = n0 + wn + j * 16 + ln;
        float v = acc[i][j][r] + bias[gn];
        if (mode == 0) {
          outF[(size_t)gm * N + gn] = v;
        } else {
          int b = gm / TSEQ, t = gm % TSEQ;
          int h = gn >> 6, dh = gn & 63;
          outH[((((size_t)b * NH + h) * TSEQ + t) << 6) + dh] =
              (_Float16)(v * scale);
        }
      }
}

// ---------------------------------------------------------------- attention
// Flash-style causal attention over [BH][T][64]. win==0 -> full causal;
// win==WIN -> sliding window (last WIN keys incl. self). Q pre-scaled 1/8.
__global__ __launch_bounds__(128) void attn_wmma(
    const _Float16* __restrict__ q, const _Float16* __restrict__ k,
    const _Float16* __restrict__ v, float* __restrict__ out, int win) {
  __shared__ _Float16 Ks[32 * 72];       // K tile, row-major [j][d]
  __shared__ _Float16 Vt[64 * 40];       // V tile transposed [d][j]
  __shared__ _Float16 Ps[4 * 16 * 40];   // per-wave P scratch
  const int tid  = threadIdx.x;
  const int lane = tid & 31;
  const int w    = tid >> 5;
  const int ln = lane & 15, lh = lane >> 4;
  const int bh = blockIdx.y;
  const int q0 = blockIdx.x * 64;
  const int qrow = q0 + w * 16;
  const size_t base = (size_t)bh * TSEQ * DHD;

  v16h aq[2];
#pragma unroll
  for (int s = 0; s < 2; ++s)
    aq[s] = ldA(&q[base + (size_t)(qrow + ln) * DHD + s * 32], lh);

  v8f o[4] = {};
  float mi[8], li[8];
#pragma unroll
  for (int r = 0; r < 8; ++r) { mi[r] = -1e30f; li[r] = 0.f; }

  int jstart = 0;
  if (win > 0) { int js = q0 - WIN; jstart = js > 0 ? js : 0; }  // mult of 32
  const int jend = q0 + 63;
  const int krow = tid >> 2;           // 0..31
  const int koff = (tid & 3) * 16;
  _Float16* Pw = &Ps[w * 16 * 40];

  for (int j0 = jstart; j0 <= jend; j0 += 32) {
    // K tile: async global->LDS; V tile: transpose on the fly (manual).
    cp_async32(&k[base + (size_t)(j0 + krow) * DHD + koff],
               &Ks[krow * 72 + koff]);
    {
      v16h vv = ldB(&v[base + (size_t)(j0 + krow) * DHD + koff]);
#pragma unroll
      for (int e = 0; e < 16; ++e) Vt[(koff + e) * 40 + krow] = vv[e];
    }
    cp_wait();
    __syncthreads();

    // hoist all K-tile B frags, then issue the 4 score WMMAs back-to-back
    v16h bk[2][2];
#pragma unroll
    for (int jf = 0; jf < 2; ++jf)
#pragma unroll
      for (int st = 0; st < 2; ++st)
        bk[jf][st] = ldB(&Ks[(jf * 16 + ln) * 72 + st * 32 + lh * 16]);
    v8f s[2] = {};
#pragma unroll
    for (int jf = 0; jf < 2; ++jf)
#pragma unroll
      for (int st = 0; st < 2; ++st)
        s[jf] = wmma16(aq[st], bk[jf][st], s[jf]);

    float corr[8];
    _Float16 p0h[8], p1h[8];
#pragma unroll
    for (int r = 0; r < 8; ++r) {
      const int t = qrow + r + 8 * lh;
      const int c0 = j0 + ln, c1 = j0 + 16 + ln;
      bool v0 = (c0 <= t) && (win == 0 || c0 > t - WIN);
      bool v1 = (c1 <= t) && (win == 0 || c1 > t - WIN);
      float s0 = v0 ? (float)s[0][r] : -1e30f;
      float s1 = v1 ? (float)s[1][r] : -1e30f;
      float mx = fmaxf(s0, s1);
#pragma unroll
      for (int d = 8; d >= 1; d >>= 1) mx = fmaxf(mx, __shfl_xor(mx, d, 32));
      float mn = fmaxf(mi[r], mx);
      float p0 = v0 ? __expf(s0 - mn) : 0.f;
      float p1 = v1 ? __expf(s1 - mn) : 0.f;
      float rs = p0 + p1;
#pragma unroll
      for (int d = 8; d >= 1; d >>= 1) rs += __shfl_xor(rs, d, 32);
      float c = __expf(mi[r] - mn);
      li[r] = li[r] * c + rs;
      mi[r] = mn;
      corr[r] = c;
      p0h[r] = (_Float16)p0;
      p1h[r] = (_Float16)p1;
    }
#pragma unroll
    for (int df = 0; df < 4; ++df)
#pragma unroll
      for (int r = 0; r < 8; ++r) o[df][r] *= corr[r];

    // C-layout -> row-major P in wave-private LDS, reload as A operand.
#pragma unroll
    for (int r = 0; r < 8; ++r) {
      Pw[(r + 8 * lh) * 40 + ln]      = p0h[r];
      Pw[(r + 8 * lh) * 40 + 16 + ln] = p1h[r];
    }
    v16h ap = ldA(&Pw[ln * 40], lh);
    v16h bv[4];
#pragma unroll
    for (int df = 0; df < 4; ++df)
      bv[df] = ldB(&Vt[(df * 16 + ln) * 40 + lh * 16]);
#pragma unroll
    for (int df = 0; df < 4; ++df) o[df] = wmma16(ap, bv[df], o[df]);
    __syncthreads();
  }

#pragma unroll
  for (int df = 0; df < 4; ++df)
#pragma unroll
    for (int r = 0; r < 8; ++r) {
      int t = qrow + r + 8 * lh;
      int d = df * 16 + ln;
      out[base + (size_t)t * DHD + d] = o[df][r] / li[r];
    }
}

// ---------------------------------------------------------------- combine
// attn[b][t][h*64+dh] = f16(full + sigmoid(gate) * windowed)
__global__ void combine_heads(const float* __restrict__ fullo,
                              const float* __restrict__ wino,
                              const float* __restrict__ gate,
                              _Float16* __restrict__ attn) {
  int i = blockIdx.x * blockDim.x + threadIdx.x;
  if (i >= NB * NH * TSEQ * DHD) return;
  float sg = 1.f / (1.f + __expf(-gate[0]));
  int dh = i & 63;
  int t  = (i >> 6) & (TSEQ - 1);
  int h  = (i >> 17) & (NH - 1);
  int b  = i >> 21;
  attn[(((size_t)b * TSEQ + t) * DMOD) + h * 64 + dh] =
      (_Float16)(fullo[i] + sg * wino[i]);
}

// ---------------------------------------------------------------- launch
extern "C" void kernel_launch(void* const* d_in, const int* in_sizes, int n_in,
                              void* d_out, int out_size, void* d_ws,
                              size_t ws_size, hipStream_t stream) {
  (void)in_sizes; (void)n_in; (void)out_size; (void)ws_size;
  const float* x    = (const float*)d_in[0];
  const float* Wq   = (const float*)d_in[1];
  const float* bq   = (const float*)d_in[2];
  const float* Wk   = (const float*)d_in[3];
  const float* bk   = (const float*)d_in[4];
  const float* Wv   = (const float*)d_in[5];
  const float* bv   = (const float*)d_in[6];
  const float* Wo   = (const float*)d_in[7];
  const float* bo   = (const float*)d_in[8];
  const float* gate = (const float*)d_in[9];

  const int M  = NB * TSEQ;       // 4096
  const int DM = DMOD;            // 1024
  char*  ws  = (char*)d_ws;
  size_t off = 0;
  auto alloc = [&](size_t bytes) -> char* {
    char* p = ws + off;
    off = (off + bytes + 255) & ~(size_t)255;
    return p;
  };
  _Float16* xh   = (_Float16*)alloc((size_t)M * DM * 2);
  _Float16* wqh  = (_Float16*)alloc((size_t)DM * DM * 2);
  _Float16* wkh  = (_Float16*)alloc((size_t)DM * DM * 2);
  _Float16* wvh  = (_Float16*)alloc((size_t)DM * DM * 2);
  _Float16* woh  = (_Float16*)alloc((size_t)DM * DM * 2);
  _Float16* qh   = (_Float16*)alloc((size_t)M * DM * 2);
  _Float16* kh   = (_Float16*)alloc((size_t)M * DM * 2);
  _Float16* vh   = (_Float16*)alloc((size_t)M * DM * 2);
  float*    fullo = (float*)alloc((size_t)M * DM * 4);
  float*    wino  = (float*)alloc((size_t)M * DM * 4);
  _Float16* attn  = (_Float16*)alloc((size_t)M * DM * 2);

  int n = M * DM;
  cvt_f32_f16<<<(n + 255) / 256, 256, 0, stream>>>(x, xh, n);
  n = DM * DM;
  cvt_f32_f16<<<(n + 255) / 256, 256, 0, stream>>>(Wq, wqh, n);
  cvt_f32_f16<<<(n + 255) / 256, 256, 0, stream>>>(Wk, wkh, n);
  cvt_f32_f16<<<(n + 255) / 256, 256, 0, stream>>>(Wv, wvh, n);
  cvt_f32_f16<<<(n + 255) / 256, 256, 0, stream>>>(Wo, woh, n);

  dim3 gB(DM / 128, M / 128);
  // Q gets 1/sqrt(DH)=0.125 folded in (bias included before scaling).
  gemm_wmma<<<gB, 256, 0, stream>>>(xh, wqh, bq, nullptr, qh, M, DM, DM, 1, 0.125f);
  gemm_wmma<<<gB, 256, 0, stream>>>(xh, wkh, bk, nullptr, kh, M, DM, DM, 1, 1.0f);
  gemm_wmma<<<gB, 256, 0, stream>>>(xh, wvh, bv, nullptr, vh, M, DM, DM, 1, 1.0f);

  dim3 gA(TSEQ / 64, NB * NH);
  attn_wmma<<<gA, 128, 0, stream>>>(qh, kh, vh, fullo, 0);
  attn_wmma<<<gA, 128, 0, stream>>>(qh, kh, vh, wino, WIN);

  n = M * DM;
  combine_heads<<<(n + 255) / 256, 256, 0, stream>>>(fullo, wino, gate, attn);

  gemm_wmma<<<gB, 256, 0, stream>>>(attn, woh, bo, (float*)d_out, nullptr, M, DM,
                                    DM, 0, 1.0f);
}
